// RNNEncoder_50002009260175
// MI455X (gfx1250) — compile-verified, matrix operand order checked by reading, not run on hardware
//
#include <hip/hip_runtime.h>

// ---------------------------------------------------------------------------
// MI455X (gfx1250, wave32, WMMA) implementation of the BAG-coupled dual LSTM
// encoder.  T=512, B=128, IN=H=OUT=128.
//
// Kernels:
//  1) convert_weights : fp32 weights -> f16, transposed to [N][K] layout.
//  2) convert_x       : fp32 a_x/v_x -> f16 (parallel; removes per-step
//                       conversion from the serial scan's critical path).
//  3) scan_kernel     : single persistent workgroup (16 waves) runs the 512
//     sequential steps with v_wmma_f32_16x16x32_f16; x and weight fragments
//     stream from L2, recurrent state lives in LDS.  Writes fp32 h to d_out
//     and an f16 copy to ws for the FC stage.
//  4) fc_kernel       : masked final FC (WMMA); stages its A-tile with
//     global_load_async_to_lds_b128 + s_wait_asynccnt (ASYNCcnt path).
//  5) gather_kernel   : bag-index gather for a_y / v_y.
// ---------------------------------------------------------------------------

typedef __attribute__((ext_vector_type(16))) _Float16 v16h;
typedef __attribute__((ext_vector_type(8)))  float    v8f;

#define T_STEPS 512
#define BATCH   128
#define HID     128
#define NW      16          // waves in scan kernel (512 threads)
#define LDA     136         // f16 row pitch in LDS (272B rows -> bank spread)
#define LD32    132         // fp32 row pitch for h_m buffer

// d_out layout: a_y | a_out | v_y | v_out  (flat fp32, reference return order)
constexpr int A_Y_OFF   = 0;
constexpr int A_OUT_OFF = BATCH * HID;                          // 16384
constexpr int V_Y_OFF   = A_OUT_OFF + BATCH * T_STEPS * HID;    // 8404992
constexpr int V_OUT_OFF = V_Y_OFF + BATCH * HID;                // 8421376

// ws layout (f16 elements)
constexpr int WS_WTA  = 0;                      // [512][256] Wcat_a^T
constexpr int WS_WTV  = WS_WTA + 512 * 256;     // [512][256] Wcat_v^T
constexpr int WS_WMB  = WS_WTV + 512 * 256;     // [128][256] W_mb^T
constexpr int WS_WB   = WS_WMB + 128 * 256;     // [128][128] W_b^T
constexpr int WS_WF   = WS_WB  + 128 * 128;     // [128][128] Wf^T
constexpr int WS_WEND = WS_WF  + 128 * 128;     // 327680
constexpr int XN      = T_STEPS * BATCH * HID;  // 8388608
constexpr int WS_XA   = WS_WEND;                // f16 a_x
constexpr int WS_XV   = WS_XA + XN;             // f16 v_x
constexpr int WS_HA16 = WS_XV + XN;             // f16 copy of a_out [B,T,H]
constexpr int WS_HV16 = WS_HA16 + XN;           // f16 copy of v_out
// total ws: 33,882,112 f16 = ~64.6 MB

// Fragment loader for v_wmma_f32_16x16x32_f16 operands.
// Source layout: row-major [16 rows][ldk] f16, rows = M (A) or N (B^T).
// Per ISA: lanes 0-15 hold row (lane&15), K in {0..7, 16..23};
//          lanes 16-31 hold K in {8..15, 24..31}.  Two 16B loads per lane.
__device__ __forceinline__ v16h frag_ld(const _Float16* p, int ldk, int lane) {
  const _Float16* q = p + (lane & 15) * ldk + ((lane >> 4) << 3);
  union { v16h v; float4 f[2]; } u;
  u.f[0] = *(const float4*)(q);
  u.f[1] = *(const float4*)(q + 16);
  return u.v;
}

__device__ __forceinline__ v8f wmma_f16(v16h a, v16h b, v8f c) {
  return __builtin_amdgcn_wmma_f32_16x16x32_f16(false, a, false, b,
                                                (short)0, c, false, false);
}

__device__ __forceinline__ float wred(float v) {
  for (int m = 16; m > 0; m >>= 1) v += __shfl_xor(v, m, 32);
  return v;
}

__device__ __forceinline__ float sigm(float x) {
  return 1.0f / (1.0f + __expf(-x));
}

// Async global->LDS copy (16B).  VDST operand is the LDS byte address; the
// low 32 bits of a flat pointer into LDS are the LDS address per the gfx1250
// aperture mapping.  Tracked by ASYNCcnt.
__device__ __forceinline__ void async_ld_b128(const _Float16* lds_ptr,
                                              const _Float16* gptr) {
  unsigned loff = (unsigned)(unsigned long long)lds_ptr;
  asm volatile("global_load_async_to_lds_b128 %0, %1, off"
               :: "v"(loff), "v"(gptr) : "memory");
}
__device__ __forceinline__ void wait_asynccnt0() {
  asm volatile("s_wait_asynccnt 0x0" ::: "memory");
}

// ---------------------------------------------------------------------------
__global__ __launch_bounds__(256) void convert_weights(
    const float* __restrict__ Wax_a, const float* __restrict__ Wah_a,
    const float* __restrict__ Wax_v, const float* __restrict__ Wah_v,
    const float* __restrict__ W_mb,  const float* __restrict__ W_b,
    const float* __restrict__ Wf,    _Float16* __restrict__ ws) {
  int i = blockIdx.x * blockDim.x + threadIdx.x;
  if (i >= WS_WEND) return;
  if (i < WS_WTV) {
    int n = i >> 8, k = i & 255;
    float v = (k < 128) ? Wax_a[k * 512 + n] : Wah_a[(k - 128) * 512 + n];
    ws[i] = (_Float16)v;
  } else if (i < WS_WMB) {
    int j = i - WS_WTV; int n = j >> 8, k = j & 255;
    float v = (k < 128) ? Wax_v[k * 512 + n] : Wah_v[(k - 128) * 512 + n];
    ws[i] = (_Float16)v;
  } else if (i < WS_WB) {
    int j = i - WS_WMB; int n = j >> 8, k = j & 255;
    ws[i] = (_Float16)W_mb[k * 128 + n];
  } else if (i < WS_WF) {
    int j = i - WS_WB; int n = j >> 7, k = j & 127;
    ws[i] = (_Float16)W_b[k * 128 + n];
  } else {
    int j = i - WS_WF; int n = j >> 7, k = j & 127;
    ws[i] = (_Float16)Wf[k * 128 + n];
  }
}

__global__ __launch_bounds__(256) void convert_x(
    const float* __restrict__ a_x, const float* __restrict__ v_x,
    _Float16* __restrict__ ws) {
  size_t i = (size_t)blockIdx.x * blockDim.x + threadIdx.x;
  if (i < (size_t)XN)            ws[WS_XA + i] = (_Float16)a_x[i];
  else if (i < 2 * (size_t)XN)   ws[WS_XA + i] = (_Float16)v_x[i - XN];
}

// ---------------------------------------------------------------------------
// LDS carve (bytes) for scan kernel; total 284160 < 320KB WGP LDS.
constexpr int SZ16   = 128 * LDA * 2;           // 34816
constexpr int OFF_HM = 0;                       // fp32 128*132*4 = 67584
constexpr int OFF_HA = 67584;
constexpr int OFF_HV = OFF_HA + SZ16;
constexpr int OFF_CA = OFF_HV + SZ16;
constexpr int OFF_CV = OFF_CA + SZ16;
constexpr int OFF_OA = OFF_CV + SZ16;
constexpr int OFF_OV = OFF_OA + SZ16;
constexpr int OFF_MA  = OFF_OV + SZ16;          // 128 f32 each
constexpr int OFF_MV  = OFF_MA + 512;
constexpr int OFF_MB  = OFF_MV + 512;
constexpr int OFF_BA  = OFF_MB + 512;           // 512 f32
constexpr int OFF_BV  = OFF_BA + 2048;          // 512 f32
constexpr int OFF_BMB = OFF_BV + 2048;          // 128 f32 each
constexpr int OFF_BB  = OFF_BMB + 512;
constexpr int OFF_LNG = OFF_BB + 512;
constexpr int OFF_LNB = OFF_LNG + 512;
constexpr int SMEM_BYTES = OFF_LNB + 512;       // 284160

__global__ __launch_bounds__(512) void scan_kernel(
    const float* __restrict__ aco, const float* __restrict__ vis,
    const float* __restrict__ bag,
    const float* __restrict__ b_a, const float* __restrict__ b_v,
    const float* __restrict__ b_mb, const float* __restrict__ b_b,
    const float* __restrict__ ln_g, const float* __restrict__ ln_b,
    _Float16* __restrict__ ws, float* __restrict__ out) {
  __shared__ __align__(16) unsigned char smem[SMEM_BYTES];
  float*    sHM = (float*)(smem + OFF_HM);
  _Float16* sHa = (_Float16*)(smem + OFF_HA);
  _Float16* sHv = (_Float16*)(smem + OFF_HV);
  _Float16* sCa = (_Float16*)(smem + OFF_CA);
  _Float16* sCv = (_Float16*)(smem + OFF_CV);
  _Float16* sOa = (_Float16*)(smem + OFF_OA);
  _Float16* sOv = (_Float16*)(smem + OFF_OV);
  float* sMa  = (float*)(smem + OFF_MA);
  float* sMv  = (float*)(smem + OFF_MV);
  float* sMb  = (float*)(smem + OFF_MB);
  float* sBa  = (float*)(smem + OFF_BA);
  float* sBv  = (float*)(smem + OFF_BV);
  float* sBmb = (float*)(smem + OFF_BMB);
  float* sBb  = (float*)(smem + OFF_BB);
  float* sLng = (float*)(smem + OFF_LNG);
  float* sLnb = (float*)(smem + OFF_LNB);

  const _Float16* WTa = ws + WS_WTA;   // [512][256]
  const _Float16* WTv = ws + WS_WTV;
  const _Float16* WMB = ws + WS_WMB;   // [128][256]
  const _Float16* WBt = ws + WS_WB;    // [128][128]
  const _Float16* xAa = ws + WS_XA;    // f16 [T][B][128]
  const _Float16* xAv = ws + WS_XV;
  _Float16* h16a = ws + WS_HA16;       // f16 [B][T][128]
  _Float16* h16v = ws + WS_HV16;
  float* a_out = out + A_OUT_OFF;
  float* v_out = out + V_OUT_OFF;

  const int tid = threadIdx.x, lane = tid & 31, wv = tid >> 5;

  // one-time staging: biases / LN params; zero h state
  if (tid < 512) { sBa[tid] = b_a[tid]; sBv[tid] = b_v[tid]; }
  if (tid < 128) {
    sBmb[tid] = b_mb[tid]; sBb[tid] = b_b[tid];
    sLng[tid] = ln_g[tid]; sLnb[tid] = ln_b[tid];
  }
  for (int e = tid; e < 128 * 128; e += 512) {
    int r = e >> 7, c = e & 127;
    sHa[r * LDA + c] = (_Float16)0.0f;
    sHv[r * LDA + c] = (_Float16)0.0f;
  }
  __syncthreads();

  for (int t = 0; t < T_STEPS; ++t) {
    // ---- stage A: masks only (x comes straight from L2-resident f16 copy)
    if (tid < 128) {
      sMa[tid] = aco[t * BATCH + tid];
      sMv[tid] = vis[t * BATCH + tid];
      sMb[tid] = bag[t * BATCH + tid];
    }
    const _Float16* xa = xAa + (size_t)t * BATCH * HID;
    const _Float16* xv = xAv + (size_t)t * BATCH * HID;
    if (t + 1 < T_STEPS) {   // warm next step's x into cache
      __builtin_prefetch(xa + BATCH * HID + tid * 32, 0, 1);
      __builtin_prefetch(xv + BATCH * HID + tid * 32, 0, 1);
    }
    __syncthreads();

    // ---- stage B: gate GEMM  ga = [x;h] @ Wcat  (i,g,o gates; f is dead)
    for (int it = wv; it < 128; it += NW) {
      int mod = it >> 6, rt = (it >> 3) & 7, ct = it & 7;
      const _Float16* xg = mod ? xv : xa;
      const _Float16* sH = mod ? sHv : sHa;
      const _Float16* WT = mod ? WTv : WTa;
      const float*    sB = mod ? sBv : sBa;
      const float*    sM = mod ? sMv : sMa;
      _Float16*       sC = mod ? sCv : sCa;
      _Float16*       sO = mod ? sOv : sOa;
      v8f accI = {}, accG = {}, accO = {};
      for (int kt = 0; kt < 8; ++kt) {
        int kk = kt * 32;
        v16h af;
        if (kk < 128) af = frag_ld(xg + rt * 16 * HID + kk, HID, lane);
        else          af = frag_ld(sH + rt * 16 * LDA + (kk - 128), LDA, lane);
        v16h bI = frag_ld(WT + (      ct * 16) * 256 + kk, 256, lane);
        v16h bG = frag_ld(WT + (256 + ct * 16) * 256 + kk, 256, lane);
        v16h bO = frag_ld(WT + (384 + ct * 16) * 256 + kk, 256, lane);
        accI = wmma_f16(af, bI, accI);
        accG = wmma_f16(af, bG, accG);
        accO = wmma_f16(af, bO, accO);
      }
      int n = lane & 15, m0 = (lane >> 4) * 8;
      for (int g = 0; g < 8; ++g) {
        int row = rt * 16 + m0 + g, col = ct * 16 + n;
        float mk = sM[row];
        float iv = accI[g] + sB[col];
        float gv = accG[g] + sB[256 + col];
        float ov = accO[g] + sB[384 + col];
        float c  = sigm(iv) * tanhf(gv) * mk;        // c_prev == 0 in reference
        float o  = sigm(ov) * mk + (1.0f - mk);
        sC[row * LDA + col] = (_Float16)c;
        sO[row * 128 + col] = (_Float16)o;
      }
    }
    __syncthreads();

    // ---- stage C: BAG + LN + output gate, both directions sequentially
    for (int dir = 0; dir < 2; ++dir) {
      const _Float16* sMain = dir ? sCv : sCa;
      const _Float16* sBr   = dir ? sCa : sCv;
      const _Float16* sOm   = dir ? sOv : sOa;
      _Float16*       sHn   = dir ? sHv : sHa;
      _Float16*       h16p  = dir ? h16v : h16a;
      float*          outp  = dir ? v_out : a_out;

      // GEMM2: wb = relu(cat(main,branch)@W_mb + b_mb); bw = branch@W_b + b_b
      for (int it = wv; it < 64; it += NW) {
        int rt = it >> 3, ct = it & 7;
        v8f acc1 = {}, acc2 = {};
        for (int kt = 0; kt < 8; ++kt) {
          int kk = kt * 32;
          v16h af;
          if (kk < 128) af = frag_ld(sMain + rt * 16 * LDA + kk, LDA, lane);
          else          af = frag_ld(sBr + rt * 16 * LDA + (kk - 128), LDA, lane);
          v16h bw = frag_ld(WMB + (ct * 16) * 256 + kk, 256, lane);
          acc1 = wmma_f16(af, bw, acc1);
        }
        for (int kt = 0; kt < 4; ++kt) {
          int kk = kt * 32;
          v16h af = frag_ld(sBr + rt * 16 * LDA + kk, LDA, lane);
          v16h bw = frag_ld(WBt + (ct * 16) * 128 + kk, 128, lane);
          acc2 = wmma_f16(af, bw, acc2);
        }
        int n = lane & 15, m0 = (lane >> 4) * 8;
        for (int g = 0; g < 8; ++g) {
          int row = rt * 16 + m0 + g, col = ct * 16 + n;
          float wb = fmaxf(acc1[g] + sBmb[col], 0.0f);
          float bv = acc2[g] + sBb[col];
          sHM[row * LD32 + col] = wb * bv;
        }
      }
      __syncthreads();

      // row-wise epilogue: alpha gate, LayerNorm, mask select, o * tanh()
      for (int rr = 0; rr < 8; ++rr) {
        int row = wv + rr * NW;
        float mv[4], hv[4];
        float em2 = 0.0f, hm2 = 0.0f;
        for (int i = 0; i < 4; ++i) {
          int col = lane + i * 32;
          float m = (float)sMain[row * LDA + col];
          float h = sHM[row * LD32 + col];
          mv[i] = m; hv[i] = h;
          em2 += m * m; hm2 += h * h;
        }
        em2 = wred(em2); hm2 = wred(hm2);
        float em = sqrtf(em2), hm = sqrtf(hm2);
        if (hm == 0.0f) hm = 1.0f;
        float alpha = fminf(em / (hm + 1e-6f), 1.0f);   // BETA_SHIFT == 1
        float xr[4], s = 0.0f;
        for (int i = 0; i < 4; ++i) { xr[i] = alpha * hv[i] + mv[i]; s += xr[i]; }
        s = wred(s);
        float mu = s * (1.0f / 128.0f), vv = 0.0f;
        for (int i = 0; i < 4; ++i) { float d = xr[i] - mu; vv += d * d; }
        vv = wred(vv);
        float inv = rsqrtf(vv * (1.0f / 128.0f) + 1e-5f);
        float mb = sMb[row];
        for (int i = 0; i < 4; ++i) {
          int col = lane + i * 32;
          float ln = (xr[i] - mu) * inv * sLng[col] + sLnb[col];
          float sc = ln * mb + mv[i] * (1.0f - mb);
          float o  = (float)sOm[row * 128 + col];
          float hn = o * tanhf(sc);
          sHn[row * LDA + col] = (_Float16)hn;                 // next-step h
          int oidx = (row * T_STEPS + t) * HID + col;          // [B,T,H]
          outp[oidx] = hn;
          h16p[oidx] = (_Float16)hn;                           // f16 for FC
        }
      }
      __syncthreads();
    }
  }
}

// ---------------------------------------------------------------------------
// masked final FC, in place over d_out's a_out / v_out regions.
// A-tile staged via async global->LDS (f16 copy written by scan_kernel).
__global__ __launch_bounds__(256) void fc_kernel(
    float* __restrict__ out, const float* __restrict__ bag,
    const float* __restrict__ bf, const _Float16* __restrict__ ws) {
  __shared__ __align__(16) _Float16 sA[64 * 128];
  int bid = blockIdx.x;
  int mod = bid >> 10;            // 1024 blocks per modality
  int blk = bid & 1023;
  float* base = out + (mod ? V_OUT_OFF : A_OUT_OFF);
  const _Float16* h16 = ws + (mod ? WS_HV16 : WS_HA16);
  const _Float16* WFt = ws + WS_WF;
  int row0 = blk * 64;
  int tid = threadIdx.x, lane = tid & 31, wv = tid >> 5;

  // async-stage 64 rows x 128 f16 = 1024 x 16B transfers (ASYNCcnt path)
  for (int e = tid; e < 1024; e += 256) {
    int off = e * 8;                       // f16 elements (8 per 16B chunk)
    async_ld_b128(sA + off, h16 + (size_t)row0 * 128 + off);
  }
  wait_asynccnt0();
  __syncthreads();

  for (int it = wv; it < 32; it += 8) {     // 4 row-tiles x 8 col-tiles
    int rt = it >> 3, ct = it & 7;
    v8f acc = {};
    for (int kt = 0; kt < 4; ++kt) {
      v16h a = frag_ld(sA + rt * 16 * 128 + kt * 32, 128, lane);
      v16h b = frag_ld(WFt + (ct * 16) * 128 + kt * 32, 128, lane);
      acc = wmma_f16(a, b, acc);
    }
    int n = lane & 15, m0 = (lane >> 4) * 8;
    for (int g = 0; g < 8; ++g) {
      int row = row0 + rt * 16 + m0 + g;
      int col = ct * 16 + n;
      int b = row >> 9, tt = row & 511;     // row = b*T + t, T = 512
      float mb = bag[tt * BATCH + b];
      float y = acc[g] + bf[col];
      float orig = base[row * 128 + col];   // exact fp32 passthrough
      base[row * 128 + col] = (mb == 1.0f) ? y : orig;
    }
  }
}

// ---------------------------------------------------------------------------
__global__ __launch_bounds__(128) void gather_kernel(
    float* __restrict__ out, const float* __restrict__ bag) {
  int b = blockIdx.x, j = threadIdx.x;
  int idx = T_STEPS - 1;
  for (int tt = T_STEPS - 1; tt >= 0; --tt) {
    if (bag[tt * BATCH + b] == 1.0f) { idx = tt; break; }
  }
  out[A_Y_OFF + b * HID + j] = out[A_OUT_OFF + (b * T_STEPS + idx) * HID + j];
  out[V_Y_OFF + b * HID + j] = out[V_OUT_OFF + (b * T_STEPS + idx) * HID + j];
}

// ---------------------------------------------------------------------------
extern "C" void kernel_launch(void* const* d_in, const int* in_sizes, int n_in,
                              void* d_out, int out_size, void* d_ws, size_t ws_size,
                              hipStream_t stream) {
  const float* a_x   = (const float*)d_in[0];
  const float* aco   = (const float*)d_in[1];
  const float* v_x   = (const float*)d_in[2];
  const float* vis   = (const float*)d_in[3];
  const float* bag   = (const float*)d_in[4];
  const float* Wax_a = (const float*)d_in[5];
  const float* Wah_a = (const float*)d_in[6];
  const float* b_a   = (const float*)d_in[7];
  const float* Wax_v = (const float*)d_in[8];
  const float* Wah_v = (const float*)d_in[9];
  const float* b_v   = (const float*)d_in[10];
  const float* W_mb  = (const float*)d_in[11];
  const float* b_mb  = (const float*)d_in[12];
  const float* W_b   = (const float*)d_in[13];
  const float* b_b   = (const float*)d_in[14];
  const float* ln_g  = (const float*)d_in[15];
  const float* ln_b  = (const float*)d_in[16];
  const float* Wf    = (const float*)d_in[17];
  const float* bf    = (const float*)d_in[18];
  float* out = (float*)d_out;
  _Float16* wsh = (_Float16*)d_ws;

  convert_weights<<<(WS_WEND + 255) / 256, 256, 0, stream>>>(
      Wax_a, Wah_a, Wax_v, Wah_v, W_mb, W_b, Wf, wsh);
  convert_x<<<(2 * XN) / 256, 256, 0, stream>>>(a_x, v_x, wsh);
  scan_kernel<<<1, 512, 0, stream>>>(
      aco, vis, bag, b_a, b_v, b_mb, b_b, ln_g, ln_b, wsh, out);
  fc_kernel<<<2048, 256, 0, stream>>>(out, bag, bf, wsh);
  gather_kernel<<<BATCH, 128, 0, stream>>>(out, bag);
}
